// LaggedSpatialAttention_32830730010787
// MI455X (gfx1250) — matrix operand order repaired and checked by reading, not compiled, not run on hardware
//
#include <hip/hip_runtime.h>

// ---------------------------------------------------------------- types
typedef __bf16 bf16_t;
typedef __attribute__((ext_vector_type(16))) __bf16 v16bf;
typedef __attribute__((ext_vector_type(8)))  __bf16 v8bf;
typedef __attribute__((ext_vector_type(8)))  float  v8f;
typedef __attribute__((ext_vector_type(4)))  unsigned int u32x4;
typedef __attribute__((ext_vector_type(8)))  int i32x8;
typedef __attribute__((ext_vector_type(4)))  int i32x4;

#define BB 16
#define NN 1024
#define LL 7
#define DD 512
#define HH 8
#define HD 64

#if defined(__HIP_DEVICE_COMPILE__) && __has_builtin(__builtin_amdgcn_tensor_load_to_lds)
#define HAVE_TDM 1
#else
#define HAVE_TDM 0
#endif

// ---------------------------------------------------------------- helpers
__device__ __forceinline__ bf16_t f2bf(float f) {
    unsigned u = __float_as_uint(f);
    u += 0x7FFFu + ((u >> 16) & 1u);           // round-to-nearest-even
    unsigned short s = (unsigned short)(u >> 16);
    bf16_t r;
    __builtin_memcpy(&r, &s, 2);
    return r;
}

__device__ __forceinline__ float bf2f(bf16_t x) {
    unsigned short s;
    __builtin_memcpy(&s, &x, 2);
    return __uint_as_float(((unsigned)s) << 16);
}

// A fragment (16x32 bf16, M x K), row-major source: lane holds row m = lane&15,
// elements 0..7 = K(h*8..+7), elements 8..15 = K(16+h*8..+7), h = lane>>4.
__device__ __forceinline__ v16bf load_a_frag(const bf16_t* base, int ld, int lane, int kbase) {
    const int m = lane & 15, hh = lane >> 4;
    const bf16_t* p = base + m * ld + kbase + hh * 8;
    union { v16bf v; v8bf h[2]; } u;
    u.h[0] = *(const v8bf*)(p);
    u.h[1] = *(const v8bf*)(p + 16);
    return u.v;
}

// B fragment (32x16 bf16, K x N) where B[k,n] = base[n*ld + k] (source rows
// are the output columns): lane n = lane&15 holds elements 0..7 = K(h*8..+7),
// elements 8..15 = K(16+h*8..+7), h = lane>>4.
__device__ __forceinline__ v16bf load_b_frag(const bf16_t* base, int ld, int lane, int kbase) {
    const int n = lane & 15, hh = lane >> 4;
    const bf16_t* p = base + n * ld + kbase + hh * 8;
    union { v16bf v; v8bf h[2]; } u;
    u.h[0] = *(const v8bf*)(p);
    u.h[1] = *(const v8bf*)(p + 16);
    return u.v;
}

__device__ __forceinline__ v8f wmma_bf16(v16bf a, v16bf b, v8f c) {
    return __builtin_amdgcn_wmma_f32_16x16x32_bf16(
        /*neg_a=*/false, a, /*neg_b=*/false, b,
        /*c_mod=*/(short)0, c, /*reuse_a=*/false, /*reuse_b=*/false);
}

// ---------------------------------------------------------------- TDM stage
// Stage a 2-D tile (rows x row_elems bf16, global row stride row_stride elems)
// into LDS (packed, ld = row_elems) with one Tensor Data Mover descriptor
// issued by wave 0.  Caller must __syncthreads() afterwards.
__device__ __forceinline__ void stage_rows_to_lds(bf16_t* lds_dst, const bf16_t* gsrc,
                                                  unsigned rows, unsigned row_elems,
                                                  unsigned row_stride, int tid, int nthreads) {
#if HAVE_TDM
    if ((tid >> 5) == 0) {
        const unsigned long long ga = (unsigned long long)(size_t)gsrc;
        const unsigned lds_addr = (unsigned)(size_t)lds_dst;   // low 32 bits = LDS offset
        u32x4 g0;
        g0[0] = 1u;                                            // count=1 (valid user D#)
        g0[1] = lds_addr;                                      // lds_addr (bytes)
        g0[2] = (unsigned)(ga & 0xFFFFFFFFu);                  // global_addr[31:0]
        g0[3] = (unsigned)((ga >> 32) & 0x1FFFFFFu) | (2u << 30); // addr[56:32] | type=2
        i32x8 g1;
        g1[0] = (int)(1u << 16);                               // wg_mask=0, data_size=1 (2B)
        g1[1] = (int)((row_elems & 0xFFFFu) << 16);            // tensor_dim0[15:0]
        g1[2] = (int)((row_elems >> 16) | (rows << 16));       // dim0[31:16] | dim1[15:0]
        g1[3] = (int)((rows >> 16) | (row_elems << 16));       // dim1[31:16] | tile_dim0
        g1[4] = (int)rows;                                     // tile_dim1 (tile_dim2=0)
        g1[5] = (int)row_stride;                               // tensor_dim0_stride[31:0]
        g1[6] = 0;                                             // stride hi / dim1_stride lo
        g1[7] = 0;
        const i32x4 zz = {0, 0, 0, 0};
#if __clang_major__ >= 23
        const i32x8 z8 = {0, 0, 0, 0, 0, 0, 0, 0};
        __builtin_amdgcn_tensor_load_to_lds(g0, g1, zz, zz, z8, 0);
#else
        __builtin_amdgcn_tensor_load_to_lds(g0, g1, zz, zz, 0);
#endif
        __builtin_amdgcn_s_wait_tensorcnt(0);
    }
#else
    const int total = (int)(rows * row_elems);
    for (int i = tid; i < total; i += nthreads) {
        const int r = i / (int)row_elems, c = i % (int)row_elems;
        lds_dst[i] = gsrc[(size_t)r * row_stride + c];
    }
#endif
}

// ---------------------------------------------------------------- kernel 1
// lag softmax + weighted aggregation + bf16 cast.  Pure HBM streaming stage.
__global__ __launch_bounds__(256) void k_prep(const float* __restrict__ cur,
                                              const float* __restrict__ lag,
                                              const float* __restrict__ lw,
                                              bf16_t* __restrict__ xcur,
                                              bf16_t* __restrict__ xagg) {
    __shared__ float w[LL];
    if (threadIdx.x == 0) {
        float mx = lw[0];
        #pragma unroll
        for (int l = 1; l < LL; ++l) mx = fmaxf(mx, lw[l]);
        float e[LL], s = 0.f;
        #pragma unroll
        for (int l = 0; l < LL; ++l) { e[l] = __expf(lw[l] - mx); s += e[l]; }
        #pragma unroll
        for (int l = 0; l < LL; ++l) w[l] = e[l] / s;
    }
    __syncthreads();
    const int i = blockIdx.x * 256 + threadIdx.x;
    const int d = i & (DD - 1);
    const int bn = i >> 9;
    xcur[i] = f2bf(cur[i]);
    const float* lp = lag + bn * (LL * DD) + d;
    float acc = 0.f;
    #pragma unroll
    for (int l = 0; l < LL; ++l) acc += lp[l * DD] * w[l];
    xagg[i] = f2bf(acc);
}

// ---------------------------------------------------------------- kernel 2
__global__ __launch_bounds__(256) void k_castw(const float* __restrict__ s0,
                                               const float* __restrict__ s1,
                                               const float* __restrict__ s2,
                                               const float* __restrict__ s3,
                                               bf16_t* __restrict__ d0,
                                               bf16_t* __restrict__ d1,
                                               bf16_t* __restrict__ d2,
                                               bf16_t* __restrict__ d3) {
    const int i = blockIdx.x * 256 + threadIdx.x;
    const float* s; bf16_t* d;
    switch (blockIdx.y) {
        case 0:  s = s0; d = d0; break;
        case 1:  s = s1; d = d1; break;
        case 2:  s = s2; d = d2; break;
        default: s = s3; d = d3; break;
    }
    d[i] = f2bf(s[i]);
}

// ---------------------------------------------------------------- kernel 3
// Q/K/V projections.  A-tile (16x512 bf16) staged into LDS by TDM; 8 waves
// each own a 16x64 output strip; B fragments software-pipelined (fully
// unrolled so the double buffer is pure register renaming).
__global__ __launch_bounds__(256) void k_qkv(const bf16_t* __restrict__ xcur,
                                             const bf16_t* __restrict__ xagg,
                                             const bf16_t* __restrict__ wqb,
                                             const bf16_t* __restrict__ wkb,
                                             const bf16_t* __restrict__ wvb,
                                             const float* __restrict__ bq,
                                             const float* __restrict__ bk,
                                             const float* __restrict__ bv,
                                             bf16_t* __restrict__ qo,
                                             bf16_t* __restrict__ ko,
                                             bf16_t* __restrict__ vt) {
    __shared__ __align__(128) bf16_t atile[16][DD];
    const int tid = threadIdx.x, lane = tid & 31, wv = tid >> 5;
    const int r0 = blockIdx.x * 16;
    const int which = blockIdx.y;
    const bf16_t* src  = (which == 0) ? xcur : xagg;
    const bf16_t* wmat = (which == 0) ? wqb : (which == 1 ? wkb : wvb);
    const float*  bias = (which == 0) ? bq  : (which == 1 ? bk  : bv);

    stage_rows_to_lds(&atile[0][0], src + (size_t)r0 * DD, 16, DD, DD, tid, 256);
    __syncthreads();

    const int c0 = wv * 64;
    const v8f z = {0.f,0.f,0.f,0.f,0.f,0.f,0.f,0.f};
    v8f acc[4] = {z, z, z, z};
    const bf16_t* wrow[4];
    #pragma unroll
    for (int t = 0; t < 4; ++t) wrow[t] = wmat + (size_t)(c0 + t * 16) * DD;

    v16bf bcur[4];
    #pragma unroll
    for (int t = 0; t < 4; ++t) bcur[t] = load_b_frag(wrow[t], DD, lane, 0);

    #pragma unroll
    for (int kc = 0; kc < 15; ++kc) {
        const int kb = kc * 32;
        v16bf a = load_a_frag(&atile[0][0], DD, lane, kb);
        v16bf bnxt[4];
        #pragma unroll
        for (int t = 0; t < 4; ++t) bnxt[t] = load_b_frag(wrow[t], DD, lane, kb + 32);
        if (kc < 12) {
            #pragma unroll
            for (int t = 0; t < 4; ++t)
                __builtin_prefetch(wrow[t] + (size_t)(lane & 15) * DD + kb + 128, 0, 3);
        }
        #pragma unroll
        for (int t = 0; t < 4; ++t) acc[t] = wmma_bf16(a, bcur[t], acc[t]);
        #pragma unroll
        for (int t = 0; t < 4; ++t) bcur[t] = bnxt[t];
    }
    {
        v16bf a = load_a_frag(&atile[0][0], DD, lane, 480);
        #pragma unroll
        for (int t = 0; t < 4; ++t) acc[t] = wmma_bf16(a, bcur[t], acc[t]);
    }

    const int n = lane & 15, hh = lane >> 4;
    #pragma unroll
    for (int t = 0; t < 4; ++t) {
        const int col = c0 + t * 16 + n;
        const float bia = bias[col];
        #pragma unroll
        for (int r = 0; r < 8; ++r) {
            const int row = r0 + r + 8 * hh;          // global row in [0, B*N)
            const float v = acc[t][r] + bia;
            if (which == 0)      qo[(size_t)row * DD + col] = f2bf(v);
            else if (which == 1) ko[(size_t)row * DD + col] = f2bf(v);
            else {
                const int b = row >> 10, m = row & (NN - 1);
                const int hsel = col >> 6, dsel = col & (HD - 1);
                vt[(((size_t)(b * HH + hsel) * HD + dsel) << 10) + m] = f2bf(v);
            }
        }
    }
}

// ---------------------------------------------------------------- kernel 4
// Fused attention per (b, h, 16-row tile): S = QK^T/8 + 0.5*adj -> softmax in
// LDS -> O = P @ V.  128 threads = 4 waves.  Q strip staged by TDM.
__global__ __launch_bounds__(128) void k_attn(const bf16_t* __restrict__ qb,
                                              const bf16_t* __restrict__ kb,
                                              const bf16_t* __restrict__ vt,
                                              const float* __restrict__ adj,
                                              bf16_t* __restrict__ ao) {
    __shared__ __align__(128) bf16_t sc[16][NN];
    __shared__ __align__(128) bf16_t qtile[16][HD];
    __shared__ float red[16][8];
    const int tid = threadIdx.x, lane = tid & 31, wv = tid >> 5;
    const int r0 = blockIdx.x * 16;
    const int b = (int)blockIdx.y >> 3, h = (int)blockIdx.y & 7;
    const int n = lane & 15, hh = lane >> 4;
    const v8f z = {0.f,0.f,0.f,0.f,0.f,0.f,0.f,0.f};

    stage_rows_to_lds(&qtile[0][0], qb + ((size_t)(b * NN + r0) * DD + h * HD),
                      16, HD, DD, tid, 128);
    __syncthreads();

    // -------- phase 1: score tile S[16, 1024].  Iterate i so the pipeline
    // guard is compile-time under full unroll (ct = wv + 4*i, 16 steps/wave).
    const v16bf a0 = load_a_frag(&qtile[0][0], HD, lane, 0);
    const v16bf a1 = load_a_frag(&qtile[0][0], HD, lane, 32);
    {
        const bf16_t* kb0 = kb + ((size_t)(b * NN + wv * 16) * DD + h * HD);
        v16bf b0 = load_b_frag(kb0, DD, lane, 0);
        v16bf b1 = load_b_frag(kb0, DD, lane, 32);
        #pragma unroll
        for (int i = 0; i < 16; ++i) {
            const int ct = wv + 4 * i;
            v16bf n0 = b0, n1 = b1;
            if (i < 15) {
                const bf16_t* kbn = kb + ((size_t)(b * NN + (ct + 4) * 16) * DD + h * HD);
                n0 = load_b_frag(kbn, DD, lane, 0);
                n1 = load_b_frag(kbn, DD, lane, 32);
            }
            v8f accs = z;
            accs = wmma_bf16(a0, b0, accs);
            accs = wmma_bf16(a1, b1, accs);
            const int m = ct * 16 + n;
            #pragma unroll
            for (int r = 0; r < 8; ++r) {
                const int lr = r + 8 * hh;
                const float v = accs[r] * 0.125f + 0.5f * adj[(size_t)(r0 + lr) * NN + m];
                sc[lr][m] = f2bf(v);
            }
            b0 = n0; b1 = n1;
        }
    }
    __syncthreads();

    // -------- phase 2: row softmax (8 threads per row)
    const int lr = tid & 15, g = tid >> 4;
    float mx = -3.0e38f;
    for (int m = g; m < NN; m += 8) mx = fmaxf(mx, bf2f(sc[lr][m]));
    red[lr][g] = mx;
    __syncthreads();
    if (g == 0) {
        float t = red[lr][0];
        #pragma unroll
        for (int j = 1; j < 8; ++j) t = fmaxf(t, red[lr][j]);
        red[lr][0] = t;
    }
    __syncthreads();
    const float rowmax = red[lr][0];
    __syncthreads();
    float s = 0.f;
    for (int m = g; m < NN; m += 8) {
        const float e = __expf(bf2f(sc[lr][m]) - rowmax);
        s += e;
        sc[lr][m] = f2bf(e);
    }
    red[lr][g] = s;
    __syncthreads();
    if (g == 0) {
        float t = 0.f;
        #pragma unroll
        for (int j = 0; j < 8; ++j) t += red[lr][j];
        red[lr][0] = t;
    }
    __syncthreads();
    const float inv = 1.0f / red[lr][0];
    for (int m = g; m < NN; m += 8) sc[lr][m] = f2bf(bf2f(sc[lr][m]) * inv);
    __syncthreads();

    // -------- phase 3: O = P @ V_head (wave owns one 16-wide d strip),
    // V fragments pipelined from global, P fragments from LDS; full unroll.
    const bf16_t* vbase = vt + ((size_t)(b * HH + h) * HD + wv * 16) * NN;
    v8f acc = z;
    v16bf bcur = load_b_frag(vbase, NN, lane, 0);
    #pragma unroll
    for (int mc = 0; mc < 31; ++mc) {
        const int kbv = mc * 32;
        v16bf af = load_a_frag(&sc[0][0], NN, lane, kbv);
        v16bf bnxt = load_b_frag(vbase, NN, lane, kbv + 32);
        acc = wmma_bf16(af, bcur, acc);
        bcur = bnxt;
    }
    {
        v16bf af = load_a_frag(&sc[0][0], NN, lane, 31 * 32);
        acc = wmma_bf16(af, bcur, acc);
    }
    const int col = h * HD + wv * 16 + n;
    #pragma unroll
    for (int r = 0; r < 8; ++r) {
        const int row = r0 + r + 8 * hh;
        ao[(size_t)(b * NN + row) * DD + col] = f2bf(acc[r]);
    }
}

// ---------------------------------------------------------------- kernel 5
// Y = AO @ Wo^T + bo ; out = LayerNorm(current + Y).  AO tile staged by TDM.
__global__ __launch_bounds__(256) void k_out(const bf16_t* __restrict__ ao,
                                             const bf16_t* __restrict__ wob,
                                             const float* __restrict__ bo,
                                             const float* __restrict__ cur,
                                             const float* __restrict__ lng,
                                             const float* __restrict__ lnb,
                                             float* __restrict__ out) {
    __shared__ __align__(128) bf16_t atile[16][DD];
    __shared__ float yt[16][DD];
    __shared__ float red[16][16];
    const int tid = threadIdx.x, lane = tid & 31, wv = tid >> 5;
    const int r0 = blockIdx.x * 16;
    const int c0 = wv * 64;
    const int n = lane & 15, hh = lane >> 4;

    stage_rows_to_lds(&atile[0][0], ao + (size_t)r0 * DD, 16, DD, DD, tid, 256);
    __syncthreads();

    const v8f z = {0.f,0.f,0.f,0.f,0.f,0.f,0.f,0.f};
    v8f acc[4] = {z, z, z, z};
    const bf16_t* wrow[4];
    #pragma unroll
    for (int t = 0; t < 4; ++t) wrow[t] = wob + (size_t)(c0 + t * 16) * DD;

    v16bf bcur[4];
    #pragma unroll
    for (int t = 0; t < 4; ++t) bcur[t] = load_b_frag(wrow[t], DD, lane, 0);

    #pragma unroll
    for (int kc = 0; kc < 15; ++kc) {
        const int kb = kc * 32;
        v16bf a = load_a_frag(&atile[0][0], DD, lane, kb);
        v16bf bnxt[4];
        #pragma unroll
        for (int t = 0; t < 4; ++t) bnxt[t] = load_b_frag(wrow[t], DD, lane, kb + 32);
        if (kc < 12) {
            #pragma unroll
            for (int t = 0; t < 4; ++t)
                __builtin_prefetch(wrow[t] + (size_t)(lane & 15) * DD + kb + 128, 0, 3);
        }
        #pragma unroll
        for (int t = 0; t < 4; ++t) acc[t] = wmma_bf16(a, bcur[t], acc[t]);
        #pragma unroll
        for (int t = 0; t < 4; ++t) bcur[t] = bnxt[t];
    }
    {
        v16bf a = load_a_frag(&atile[0][0], DD, lane, 480);
        #pragma unroll
        for (int t = 0; t < 4; ++t) acc[t] = wmma_bf16(a, bcur[t], acc[t]);
    }

    #pragma unroll
    for (int t = 0; t < 4; ++t) {
        const int col = c0 + t * 16 + n;
        const float bia = bo[col];
        #pragma unroll
        for (int r = 0; r < 8; ++r) yt[r + 8 * hh][col] = acc[t][r] + bia;
    }
    __syncthreads();

    // LayerNorm with residual: 16 threads per row
    const int lr = tid & 15, g = tid >> 4;
    const int grow = r0 + lr;
    float s = 0.f;
    for (int m = g; m < DD; m += 16) {
        const float x = cur[(size_t)grow * DD + m] + yt[lr][m];
        yt[lr][m] = x;
        s += x;
    }
    red[lr][g] = s;
    __syncthreads();
    if (g == 0) {
        float t = 0.f;
        #pragma unroll
        for (int j = 0; j < 16; ++j) t += red[lr][j];
        red[lr][0] = t;
    }
    __syncthreads();
    const float mu = red[lr][0] * (1.0f / (float)DD);
    __syncthreads();
    float v = 0.f;
    for (int m = g; m < DD; m += 16) {
        const float d = yt[lr][m] - mu;
        v += d * d;
    }
    red[lr][g] = v;
    __syncthreads();
    if (g == 0) {
        float t = 0.f;
        #pragma unroll
        for (int j = 0; j < 16; ++j) t += red[lr][j];
        red[lr][0] = t;
    }
    __syncthreads();
    const float rstd = rsqrtf(red[lr][0] * (1.0f / (float)DD) + 1e-5f);
    for (int m = g; m < DD; m += 16)
        out[(size_t)grow * DD + m] = (yt[lr][m] - mu) * rstd * lng[m] + lnb[m];
}

// ---------------------------------------------------------------- launcher
extern "C" void kernel_launch(void* const* d_in, const int* in_sizes, int n_in,
                              void* d_out, int out_size, void* d_ws, size_t ws_size,
                              hipStream_t stream) {
    const float* cur = (const float*)d_in[0];
    const float* lag = (const float*)d_in[1];
    const float* lw  = (const float*)d_in[2];
    const float* Wq  = (const float*)d_in[3];
    const float* bq  = (const float*)d_in[4];
    const float* Wk  = (const float*)d_in[5];
    const float* bk  = (const float*)d_in[6];
    const float* Wv  = (const float*)d_in[7];
    const float* bv  = (const float*)d_in[8];
    const float* Wo  = (const float*)d_in[9];
    const float* bo  = (const float*)d_in[10];
    const float* adj = (const float*)d_in[11];
    const float* lng = (const float*)d_in[12];
    const float* lnb = (const float*)d_in[13];
    float* out = (float*)d_out;

    const size_t BND = (size_t)BB * NN * DD;   // 8,388,608
    const size_t WSZ = (size_t)DD * DD;        // 262,144
    char* p = (char*)d_ws;
    bf16_t* xcur = (bf16_t*)p; p += BND * 2;
    bf16_t* xagg = (bf16_t*)p; p += BND * 2;
    bf16_t* wqb  = (bf16_t*)p; p += WSZ * 2;
    bf16_t* wkb  = (bf16_t*)p; p += WSZ * 2;
    bf16_t* wvb  = (bf16_t*)p; p += WSZ * 2;
    bf16_t* wob  = (bf16_t*)p; p += WSZ * 2;
    bf16_t* qb   = (bf16_t*)p; p += BND * 2;
    bf16_t* kbuf = (bf16_t*)p; p += BND * 2;
    bf16_t* vt   = (bf16_t*)p; p += BND * 2;
    bf16_t* ao   = (bf16_t*)p; p += BND * 2;

    k_prep<<<dim3((unsigned)(BND / 256)), 256, 0, stream>>>(cur, lag, lw, xcur, xagg);
    k_castw<<<dim3((unsigned)(WSZ / 256), 4), 256, 0, stream>>>(Wq, Wk, Wv, Wo,
                                                               wqb, wkb, wvb, wob);
    k_qkv<<<dim3(BB * NN / 16, 3), 256, 0, stream>>>(xcur, xagg, wqb, wkb, wvb,
                                                     bq, bk, bv, qb, kbuf, vt);
    k_attn<<<dim3(NN / 16, BB * HH), 128, 0, stream>>>(qb, kbuf, vt, adj, ao);
    k_out<<<dim3(BB * NN / 16), 256, 0, stream>>>(ao, wob, bo, cur, lng, lnb, out);
}
